// MultiHeadAttentionWithRelativePE_72739566125202
// MI455X (gfx1250) — matrix operand order repaired and checked by reading, not compiled, not run on hardware
//
#include <hip/hip_runtime.h>
#include <cstdint>

// ---------------- CDNA5 WMMA types / fragments ----------------
typedef __attribute__((ext_vector_type(16))) __bf16 v16bf;
typedef __attribute__((ext_vector_type(8)))  float  v8f;
typedef __attribute__((ext_vector_type(4)))  unsigned int v4u;

union Frag {               // 8 VGPRs: A (16x32 bf16) or B (32x16 bf16)
  v16bf  v;
  __bf16 h[16];
  v4u    q2[2];
};

__device__ __forceinline__ v8f wmma_bf16(const Frag& a, const Frag& b, v8f c) {
  // D = A(16x32) * B(32x16) + C, fp32 accumulate
  return __builtin_amdgcn_wmma_f32_16x16x32_bf16(false, a.v, false, b.v,
                                                 (short)0, c, false, false);
}

// A-matrix 16x32 bf16 (ISA 7.12.2): lanes 0-15 rows M=0..15 hold K=[k0,k0+8)+[k0+16,k0+24);
// lanes 16-31 hold K offset +8. Two 16B vector loads per lane.
__device__ __forceinline__ void load_a_rm(Frag& a, const __bf16* base, int ld,
                                          int row0, int k0, int lane) {
  const int r  = row0 + (lane & 15);
  const int kb = k0 + ((lane >> 4) << 3);
  const __bf16* p = base + r * ld + kb;
  a.q2[0] = *reinterpret_cast<const v4u*>(p);
  a.q2[1] = *reinterpret_cast<const v4u*>(p + 16);
}

// B-matrix 32x16 bf16 from K-contiguous ("column-major") storage Bt[n][k]:
// lane holds column n = col0 + (lane&15), K = k0 + (lane>=16 ? 8 : 0) + {0..7, 16..23},
// matching the A layout -> two 16B vector loads.
__device__ __forceinline__ void load_b_cm(Frag& b, const __bf16* baseT, int ld,
                                          int k0, int col0, int lane) {
  const int col = col0 + (lane & 15);
  const int kb  = k0 + ((lane >> 4) << 3);
  const __bf16* p = baseT + col * ld + kb;
  b.q2[0] = *reinterpret_cast<const v4u*>(p);
  b.q2[1] = *reinterpret_cast<const v4u*>(p + 16);
}

// ---------------- prep ----------------
__global__ void cvt_f32_to_bf16_kernel(const float* __restrict__ s,
                                       __bf16* __restrict__ d, int n) {
  int i = blockIdx.x * blockDim.x + threadIdx.x;
  if (i < n) d[i] = (__bf16)s[i];
}

// dT[n][k] = s[k][n] for a 768x768 matrix (weight -> B-operand layout, bf16)
__global__ void cvt_transpose_bf16_kernel(const float* __restrict__ s,
                                          __bf16* __restrict__ dT) {
  int tid = blockIdx.x * blockDim.x + threadIdx.x;   // 768*768
  int k = tid % 768, n = tid / 768;
  dT[tid] = (__bf16)s[k * 768 + n];
}

// RT[i][k][c] = rel_pos[i - k + 63][c]  (K=c contiguous => B-operand layout;
// contiguous row copies, fully coalesced)
__global__ void gather_relpos_kernel(const float* __restrict__ rp,
                                     __bf16* __restrict__ RT) {
  int tid = blockIdx.x * blockDim.x + threadIdx.x;   // 64*64*64
  int c = tid & 63, k = (tid >> 6) & 63, i = tid >> 12;
  RT[tid] = (__bf16)rp[(i - k + 63) * 64 + c];
}

// ---------------- QKV projection: Y = X @ W + b ----------------
// grid (256 Mtiles, 12 heads, 3={q,k,v}); 1 wave/block computes 16x64 tile.
// Q,K stored row-major per head; V stored transposed (Vt[head][d][pos]).
__global__ void qkv_gemm_kernel(const __bf16* __restrict__ xb,
                                const __bf16* __restrict__ wqT,
                                const __bf16* __restrict__ wkT,
                                const __bf16* __restrict__ wvT,
                                const float* __restrict__ bq,
                                const float* __restrict__ bk,
                                const float* __restrict__ bv,
                                __bf16* __restrict__ qout,   // [12][4096][64]
                                __bf16* __restrict__ kout,   // [12][4096][64]
                                __bf16* __restrict__ vT) {   // [12][64][4096]
  const int mt = blockIdx.x, head = blockIdx.y, which = blockIdx.z;
  const int lane = threadIdx.x, hlf = lane >> 4, nlo = lane & 15;
  const __bf16* WT  = (which == 0) ? wqT : (which == 1) ? wkT : wvT;
  const float* bias = (which == 0) ? bq  : (which == 1) ? bk  : bv;
  v8f acc[4];
#pragma unroll
  for (int i = 0; i < 4; ++i) { v8f z = {}; acc[i] = z; }
  for (int k0 = 0; k0 < 768; k0 += 32) {
    Frag a; load_a_rm(a, xb, 768, mt * 16, k0, lane);
#pragma unroll
    for (int nt = 0; nt < 4; ++nt) {
      Frag b; load_b_cm(b, WT, 768, k0, head * 64 + nt * 16, lane);
      acc[nt] = wmma_bf16(a, b, acc[nt]);
    }
  }
#pragma unroll
  for (int nt = 0; nt < 4; ++nt) {
    const int d = nt * 16 + nlo;
    const float bb = bias[head * 64 + d];
#pragma unroll
    for (int r = 0; r < 8; ++r) {
      const int p = mt * 16 + r + 8 * hlf;
      const float y = acc[nt][r] + bb;
      if (which == 0)      qout[(head * 4096 + p) * 64 + d] = (__bf16)y;
      else if (which == 1) kout[(head * 4096 + p) * 64 + d] = (__bf16)y;
      else                 vT[(head * 64 + d) * 4096 + p]   = (__bf16)y;  // transposed
    }
  }
}

// ---------------- relative-position bias GEMMs ----------------
// mode 0: idx=qh: rel_h[head][qh][w][k] = sum_c q[head][qh*64+w][c]*Rh[qh][k][c]
// mode 1: idx=w : rel_w[head][qh][w][k] = sum_c q[head][qh*64+w][c]*Rw[w][k][c]
__global__ void rel_gemm_kernel(const __bf16* __restrict__ qm,
                                const __bf16* __restrict__ Rh,
                                const __bf16* __restrict__ Rw,
                                float* __restrict__ rel_h,
                                float* __restrict__ rel_w) {
  const int head = blockIdx.x, idx = blockIdx.y, mode = blockIdx.z;
  const int lane = threadIdx.x, hlf = lane >> 4, nlo = lane & 15;
  const __bf16* A; int lda; const __bf16* Bt;
  if (mode == 0) { A = qm + (head * 4096 + idx * 64) * 64; lda = 64;   Bt = Rh + idx * 4096; }
  else           { A = qm + (head * 4096 + idx) * 64;      lda = 4096; Bt = Rw + idx * 4096; }
  for (int mt = 0; mt < 4; ++mt) {
    Frag a0, a1;
    load_a_rm(a0, A, lda, mt * 16, 0,  lane);
    load_a_rm(a1, A, lda, mt * 16, 32, lane);
#pragma unroll
    for (int nt = 0; nt < 4; ++nt) {
      Frag b; v8f acc = {};
      load_b_cm(b, Bt, 64, 0,  nt * 16, lane);
      acc = wmma_bf16(a0, b, acc);
      load_b_cm(b, Bt, 64, 32, nt * 16, lane);
      acc = wmma_bf16(a1, b, acc);
#pragma unroll
      for (int r = 0; r < 8; ++r) {
        const int row = mt * 16 + r + 8 * hlf;
        const int k   = nt * 16 + nlo;
        if (mode == 0) rel_h[((head * 64 + idx) * 64 + row) * 64 + k] = acc[r];
        else           rel_w[((head * 64 + row) * 64 + idx) * 64 + k] = acc[r];
      }
    }
  }
}

// ---------------- fused flash attention w/ rel-pos bias ----------------
// grid (12 heads, 256 query tiles of 16); 1 wave/block. Online softmax.
__global__ void flash_attn_kernel(const __bf16* __restrict__ qm,   // [12][4096][64]
                                  const __bf16* __restrict__ km,   // [12][4096][64]
                                  const __bf16* __restrict__ vT,   // [12][64][4096]
                                  const float* __restrict__ rel_h,
                                  const float* __restrict__ rel_w,
                                  __bf16* __restrict__ ocat) {     // [4096][768]
  __shared__ float  hb[16][64];      // rel_h bias for this q-tile: [m][kh]
  __shared__ float  wb[16][64];      // rel_w bias for this q-tile: [m][kw]
  __shared__ __bf16 ptile[16][32];   // P tile staging (C-layout -> A-layout)
  const int head = blockIdx.x, qt = blockIdx.y;
  const int lane = threadIdx.x, hlf = lane >> 4, nlo = lane & 15;
  const int qhh = qt >> 2;           // all 16 queries share qh
  const int qw0 = (qt & 3) << 4;     // base qw of this tile

  for (int i = lane; i < 16 * 64; i += 32) {
    const int m = i >> 6, k = i & 63;
    const int base = ((head * 64 + qhh) * 64 + (qw0 + m)) * 64 + k;
    hb[m][k] = rel_h[base];
    wb[m][k] = rel_w[base];
  }
  __syncthreads();

  Frag a0, a1;  // Q tile (16x64) as two A fragments (K=0..31, 32..63)
  const __bf16* qbase = qm + (head * 4096 + qt * 16) * 64;
  load_a_rm(a0, qbase, 64, 0, 0,  lane);
  load_a_rm(a1, qbase, 64, 0, 32, lane);

  v8f o[4];
#pragma unroll
  for (int i = 0; i < 4; ++i) { v8f z = {}; o[i] = z; }
  float rmax[8], rsum[8];
#pragma unroll
  for (int r = 0; r < 8; ++r) { rmax[r] = -3.0e38f; rsum[r] = 0.0f; }

  const __bf16* ktb = km + head * 4096 * 64;   // row-major [key][c] => B for S
  const __bf16* vtb = vT + head * 64 * 4096;   // [d][key]           => B for P@V

  for (int key0 = 0; key0 < 4096; key0 += 32) {   // 32 keys / iter
    const int kh  = key0 >> 6;                    // constant over the 32 keys
    const int kwb = key0 & 63;
    v8f s0 = {}, s1 = {};
    {
      Frag b;
      load_b_cm(b, ktb, 64, 0,  key0,      lane); s0 = wmma_bf16(a0, b, s0);
      load_b_cm(b, ktb, 64, 32, key0,      lane); s0 = wmma_bf16(a1, b, s0);
      load_b_cm(b, ktb, 64, 0,  key0 + 16, lane); s1 = wmma_bf16(a0, b, s1);
      load_b_cm(b, ktb, 64, 32, key0 + 16, lane); s1 = wmma_bf16(a1, b, s1);
    }
    float fac[8];
#pragma unroll
    for (int r = 0; r < 8; ++r) {
      const int m = r + 8 * hlf;
      const float h0 = hb[m][kh];
      const float x0 = s0[r] * 8.0f + h0 + wb[m][kwb + nlo];        // scale=sqrt(64)
      const float x1 = s1[r] * 8.0f + h0 + wb[m][kwb + 16 + nlo];
      float mx = fmaxf(x0, x1);
#pragma unroll
      for (int off = 1; off < 16; off <<= 1) mx = fmaxf(mx, __shfl_xor(mx, off, 32));
      const float mn = fmaxf(rmax[r], mx);
      fac[r]  = __expf(rmax[r] - mn);
      rmax[r] = mn;
      const float p0 = __expf(x0 - mn);
      const float p1 = __expf(x1 - mn);
      float rs = p0 + p1;
#pragma unroll
      for (int off = 1; off < 16; off <<= 1) rs += __shfl_xor(rs, off, 32);
      rsum[r] = rsum[r] * fac[r] + rs;
      ptile[m][nlo]      = (__bf16)p0;
      ptile[m][16 + nlo] = (__bf16)p1;
    }
    __syncthreads();
#pragma unroll
    for (int nt = 0; nt < 4; ++nt)
#pragma unroll
      for (int r = 0; r < 8; ++r) o[nt][r] *= fac[r];
    Frag pa;   // P (16x32) re-read from LDS in A-matrix layout
    pa.q2[0] = *reinterpret_cast<const v4u*>(&ptile[nlo][hlf * 8]);
    pa.q2[1] = *reinterpret_cast<const v4u*>(&ptile[nlo][hlf * 8 + 16]);
#pragma unroll
    for (int nt = 0; nt < 4; ++nt) {
      Frag bv;
      load_b_cm(bv, vtb, 4096, key0, nt * 16, lane);
      o[nt] = wmma_bf16(pa, bv, o[nt]);
    }
    __syncthreads();
  }
#pragma unroll
  for (int nt = 0; nt < 4; ++nt) {
    const int c = head * 64 + nt * 16 + nlo;
#pragma unroll
    for (int r = 0; r < 8; ++r) {
      const int p = qt * 16 + r + 8 * hlf;
      ocat[p * 768 + c] = (__bf16)(o[nt][r] / rsum[r]);
    }
  }
}

// ---------------- output projection: out = ocat @ wo + bo ----------------
__global__ void proj_gemm_kernel(const __bf16* __restrict__ ocat,
                                 const __bf16* __restrict__ woT,
                                 const float* __restrict__ bo,
                                 float* __restrict__ out) {
  const int mt = blockIdx.x, nb = blockIdx.y;
  const int lane = threadIdx.x, hlf = lane >> 4, nlo = lane & 15;
  v8f acc[4];
#pragma unroll
  for (int i = 0; i < 4; ++i) { v8f z = {}; acc[i] = z; }
  for (int k0 = 0; k0 < 768; k0 += 32) {
    Frag a; load_a_rm(a, ocat, 768, mt * 16, k0, lane);
#pragma unroll
    for (int nt = 0; nt < 4; ++nt) {
      Frag b; load_b_cm(b, woT, 768, k0, nb * 64 + nt * 16, lane);
      acc[nt] = wmma_bf16(a, b, acc[nt]);
    }
  }
#pragma unroll
  for (int nt = 0; nt < 4; ++nt) {
    const int c = nb * 64 + nt * 16 + nlo;
    const float bb = bo[c];
#pragma unroll
    for (int r = 0; r < 8; ++r) {
      const int p = mt * 16 + r + 8 * hlf;
      out[p * 768 + c] = acc[nt][r] + bb;
    }
  }
}

// ---------------- workspace layout ----------------
constexpr size_t SZ_XB  = (size_t)4096 * 768 * 2;
constexpr size_t SZ_W   = (size_t)768 * 768 * 2;
constexpr size_t SZ_RT  = (size_t)64 * 64 * 64 * 2;
constexpr size_t SZ_QKV = (size_t)12 * 4096 * 64 * 2;
constexpr size_t SZ_REL = (size_t)12 * 64 * 64 * 64 * 4;
constexpr size_t OFF_XB   = 0;
constexpr size_t OFF_WQ   = OFF_XB + SZ_XB;
constexpr size_t OFF_WK   = OFF_WQ + SZ_W;
constexpr size_t OFF_WV   = OFF_WK + SZ_W;
constexpr size_t OFF_WO   = OFF_WV + SZ_W;
constexpr size_t OFF_RH   = OFF_WO + SZ_W;
constexpr size_t OFF_RW   = OFF_RH + SZ_RT;
constexpr size_t OFF_Q    = OFF_RW + SZ_RT;
constexpr size_t OFF_K    = OFF_Q + SZ_QKV;
constexpr size_t OFF_VT   = OFF_K + SZ_QKV;
constexpr size_t OFF_RELH = OFF_VT + SZ_QKV;
constexpr size_t OFF_RELW = OFF_RELH + SZ_REL;
constexpr size_t OFF_OC   = OFF_RELW + SZ_REL;

extern "C" void kernel_launch(void* const* d_in, const int* in_sizes, int n_in,
                              void* d_out, int out_size, void* d_ws, size_t ws_size,
                              hipStream_t stream) {
  (void)in_sizes; (void)n_in; (void)out_size; (void)ws_size;
  const float* x   = (const float*)d_in[0];
  const float* wq  = (const float*)d_in[1];
  const float* bq  = (const float*)d_in[2];
  const float* wk  = (const float*)d_in[3];
  const float* bk  = (const float*)d_in[4];
  const float* wv  = (const float*)d_in[5];
  const float* bv  = (const float*)d_in[6];
  const float* wo  = (const float*)d_in[7];
  const float* bo  = (const float*)d_in[8];
  const float* rph = (const float*)d_in[9];
  const float* rpw = (const float*)d_in[10];
  float* out = (float*)d_out;

  char* ws = (char*)d_ws;
  __bf16* xb   = (__bf16*)(ws + OFF_XB);
  __bf16* wqT  = (__bf16*)(ws + OFF_WQ);
  __bf16* wkT  = (__bf16*)(ws + OFF_WK);
  __bf16* wvT  = (__bf16*)(ws + OFF_WV);
  __bf16* woT  = (__bf16*)(ws + OFF_WO);
  __bf16* Rh   = (__bf16*)(ws + OFF_RH);
  __bf16* Rw   = (__bf16*)(ws + OFF_RW);
  __bf16* qbuf = (__bf16*)(ws + OFF_Q);
  __bf16* kbuf = (__bf16*)(ws + OFF_K);
  __bf16* vTb  = (__bf16*)(ws + OFF_VT);
  float*  relh = (float*)(ws + OFF_RELH);
  float*  relw = (float*)(ws + OFF_RELW);
  __bf16* ocat = (__bf16*)(ws + OFF_OC);

  // 1. convert inputs to bf16 (weights transposed to B-operand layout)
  cvt_f32_to_bf16_kernel<<<(3145728 + 255) / 256, 256, 0, stream>>>(x, xb, 3145728);
  cvt_transpose_bf16_kernel<<<2304, 256, 0, stream>>>(wq, wqT);
  cvt_transpose_bf16_kernel<<<2304, 256, 0, stream>>>(wk, wkT);
  cvt_transpose_bf16_kernel<<<2304, 256, 0, stream>>>(wv, wvT);
  cvt_transpose_bf16_kernel<<<2304, 256, 0, stream>>>(wo, woT);
  gather_relpos_kernel<<<1024, 256, 0, stream>>>(rph, Rh);
  gather_relpos_kernel<<<1024, 256, 0, stream>>>(rpw, Rw);

  // 2. QKV projections (V stored transposed per head)
  qkv_gemm_kernel<<<dim3(256, 12, 3), 32, 0, stream>>>(
      xb, wqT, wkT, wvT, bq, bk, bv, qbuf, kbuf, vTb);

  // 3. relative-position bias tables
  rel_gemm_kernel<<<dim3(12, 64, 2), 32, 0, stream>>>(qbuf, Rh, Rw, relh, relw);

  // 4. fused flash attention
  flash_attn_kernel<<<dim3(12, 256), 32, 0, stream>>>(qbuf, kbuf, vTb, relh, relw, ocat);

  // 5. output projection
  proj_gemm_kernel<<<dim3(256, 12), 32, 0, stream>>>(ocat, woT, bo, out);
}
